// DP2Net_8280696947091
// MI455X (gfx1250) — matrix operand order repaired and checked by reading, not compiled, not run on hardware
//
#include <hip/hip_runtime.h>

#define D      128
#define DIN    512
#define DH     256
#define NU     50000
#define NO     100000
#define NI     200000
#define NCNT   20000
#define NNODES 150000
#define NNZ_E  4000000
#define LAYERS 3

typedef __attribute__((ext_vector_type(16))) __bf16       v16bf;
typedef __attribute__((ext_vector_type(8)))  float        v8f;
typedef __attribute__((ext_vector_type(8)))  unsigned int v8u;

__device__ __forceinline__ __bf16 f2bf(float f) {
    unsigned u; __builtin_memcpy(&u, &f, 4);
    unsigned r = u + 0x7fffu + ((u >> 16) & 1u);     // round-to-nearest-even
    unsigned short s = (unsigned short)(r >> 16);
    __bf16 h; __builtin_memcpy(&h, &s, 2); return h;
}
__device__ __forceinline__ float bf2f(__bf16 h) {
    unsigned short s; __builtin_memcpy(&s, &h, 2);
    unsigned u = ((unsigned)s) << 16;
    float f; __builtin_memcpy(&f, &u, 4); return f;
}
__device__ __forceinline__ float lrelu(float x) { return x > 0.f ? x : 0.01f * x; }

// ---- CDNA5 async copy: global -> LDS, 16 bytes per lane (ASYNCcnt tracked) ----
__device__ __forceinline__ void async_ld_b128(void* lds, const void* g) {
    unsigned off = (unsigned)(unsigned long long)lds;   // low 32 bits = LDS offset
    asm volatile("global_load_async_to_lds_b128 %0, %1, off"
                 :: "v"(off), "v"(g) : "memory");
}
__device__ __forceinline__ void wait_async0() {
    asm volatile("s_wait_asynccnt 0x0" ::: "memory");
}

// ---- WMMA fragment loaders (contiguous k = 16*(lane/16) + s, same order in A and B) ----
// A: 16x32 bf16, row = lane%16; VGPR v: K=16g+2v  (g=lane/16)
__device__ __forceinline__ v16bf load_a_frag(const __bf16* base, int lda, int lane) {
    const int g = lane >> 4, r = lane & 15;
    const unsigned* p = (const unsigned*)(base + r * lda + 16 * g);
    const unsigned* q = (const unsigned*)(base + r * lda + 16 * g + 8);
    v8u u;
    u[0] = p[0]; u[1] = p[1]; u[2] = p[2]; u[3] = p[3];
    u[4] = q[0]; u[5] = q[1]; u[6] = q[2]; u[7] = q[3];
    return __builtin_bit_cast(v16bf, u);
}
// B from TRANSPOSED weight tile sWt[n][k] (ldk stride): lanes 0-15 K=0..15, lanes 16-31 K=16..31
__device__ __forceinline__ v16bf load_b_frag(const __bf16* base, int ldk, int lane) {
    const int g = lane >> 4, c = lane & 15;
    const unsigned* p = (const unsigned*)(base + c * ldk + 16 * g);
    v8u u;
#pragma unroll
    for (int i = 0; i < 8; ++i) u[i] = p[i];
    return __builtin_bit_cast(v16bf, u);
}
#define WMMA_BF16(a, b, c) \
    __builtin_amdgcn_wmma_f32_16x16x32_bf16(false, (a), false, (b), (short)0, (c), false, false)

// ---------------- small utility kernels ----------------
__global__ __launch_bounds__(256) void wt_kernel(const float* __restrict__ w,
                                                 __bf16* __restrict__ wt, int K, int Nc) {
    int i = blockIdx.x * 256 + threadIdx.x;
    if (i < K * Nc) {
        int k = i / Nc, n = i - k * Nc;
        wt[(size_t)n * K + k] = f2bf(w[(size_t)k * Nc + n]);
    }
}

__global__ __launch_bounds__(256) void zero_kernel(float* __restrict__ p, int n4) {
    int i = blockIdx.x * 256 + threadIdx.x;
    if (i < n4) { float4 z; z.x = z.y = z.z = z.w = 0.f; ((float4*)p)[i] = z; }
}

// ---------------- fused 2-layer MLP:  Y = lrelu(X@W1+b1) @ W2 + b2 ----------------
__global__ __launch_bounds__(256)
void mlp_kernel(const float* __restrict__ X, const __bf16* __restrict__ W1t,
                const float* __restrict__ b1, const __bf16* __restrict__ W2t,
                const float* __restrict__ b2, void* __restrict__ Yout,
                int M, int out_bf16) {
    __shared__ __align__(16) __bf16 sA[64 * 34];     // A K-chunk (bf16)
    __shared__ __align__(16) __bf16 sB[256 * 40];    // full-width B K-slice (16B-aligned rows)
    __shared__ __align__(16) __bf16 sH[64 * 264];    // hidden (bf16) 64 x 256

    const int tid = threadIdx.x, lane = tid & 31, wave = tid >> 5;
    const int wr = wave >> 1, wc = wave & 1;         // 4 (M) x 2 (N within 64-col group)
    const int g = lane >> 4, cidx = lane & 15;
    const int r0 = blockIdx.x * 64;
    const int arow = tid >> 2, akk = (tid & 3) * 8;
    const size_t xrow = (size_t)min(r0 + arow, M - 1);

    // -------- stage 1: hidden = lrelu(X @ W1 + b1), N=256, A staged ONCE per K-step --------
    v8f acc[4][2];
#pragma unroll
    for (int nb = 0; nb < 4; ++nb)
#pragma unroll
        for (int t = 0; t < 2; ++t) acc[nb][t] = (v8f){0,0,0,0,0,0,0,0};

    for (int k0 = 0; k0 < DIN; k0 += 32) {
        {   // A: fp32 -> bf16 convert (VALU co-executes with WMMA)
            const float* src = X + xrow * DIN + k0 + akk;
            float4 f0 = ((const float4*)src)[0];
            float4 f1 = ((const float4*)src)[1];
            __bf16* da = sA + arow * 34 + akk;
            da[0] = f2bf(f0.x); da[1] = f2bf(f0.y); da[2] = f2bf(f0.z); da[3] = f2bf(f0.w);
            da[4] = f2bf(f1.x); da[5] = f2bf(f1.y); da[6] = f2bf(f1.z); da[7] = f2bf(f1.w);
        }
        {   // B: async copy, thread t owns row n=t (64B of K-slice)
            const __bf16* gsrc = W1t + (size_t)tid * DIN + k0;
            __bf16* dst = sB + tid * 40;
#pragma unroll
            for (int q = 0; q < 4; ++q) async_ld_b128(dst + q * 8, gsrc + q * 8);
        }
        wait_async0();
        __syncthreads();
        v16bf a = load_a_frag(sA + wr * 16 * 34, 34, lane);
#pragma unroll
        for (int nb = 0; nb < 4; ++nb)
#pragma unroll
            for (int t = 0; t < 2; ++t) {
                v16bf bb = load_b_frag(sB + (nb * 64 + wc * 32 + t * 16) * 40, 40, lane);
                acc[nb][t] = WMMA_BF16(a, bb, acc[nb][t]);
            }
        __syncthreads();
    }
#pragma unroll
    for (int nb = 0; nb < 4; ++nb)
#pragma unroll
        for (int t = 0; t < 2; ++t) {
            int col = nb * 64 + wc * 32 + t * 16 + cidx;
            float bias = b1[col];
#pragma unroll
            for (int v = 0; v < 8; ++v) {
                int row = wr * 16 + v + 8 * g;
                sH[row * 264 + col] = f2bf(lrelu(acc[nb][t][v] + bias));
            }
        }
    // -------- stage 2: Y = hidden @ W2 + b2, N=128 --------
    v8f acc2[2][2];
#pragma unroll
    for (int nb = 0; nb < 2; ++nb)
#pragma unroll
        for (int t = 0; t < 2; ++t) acc2[nb][t] = (v8f){0,0,0,0,0,0,0,0};

    for (int k0 = 0; k0 < DH; k0 += 32) {
        {   // B: 128 rows, 2 threads per row
            int row = tid >> 1, half = tid & 1;
            const __bf16* gsrc = W2t + (size_t)row * DH + k0 + half * 16;
            __bf16* dst = sB + row * 40 + half * 16;
            async_ld_b128(dst, gsrc);
            async_ld_b128(dst + 8, gsrc + 8);
        }
        wait_async0();
        __syncthreads();
        v16bf a = load_a_frag(sH + wr * 16 * 264 + k0, 264, lane);
#pragma unroll
        for (int nb = 0; nb < 2; ++nb)
#pragma unroll
            for (int t = 0; t < 2; ++t) {
                v16bf bb = load_b_frag(sB + (nb * 64 + wc * 32 + t * 16) * 40, 40, lane);
                acc2[nb][t] = WMMA_BF16(a, bb, acc2[nb][t]);
            }
        __syncthreads();
    }
#pragma unroll
    for (int nb = 0; nb < 2; ++nb)
#pragma unroll
        for (int t = 0; t < 2; ++t) {
            int col = nb * 64 + wc * 32 + t * 16 + cidx;
            float bias = b2[col];
#pragma unroll
            for (int v = 0; v < 8; ++v) {
                int row = r0 + wr * 16 + v + 8 * g;
                if (row < M) {
                    float val = acc2[nb][t][v] + bias;
                    if (out_bf16) ((__bf16*)Yout)[(size_t)row * D + col] = f2bf(val);
                    else          ((float*)Yout)[(size_t)row * D + col] = val;
                }
            }
        }
}

// ---------------- fused I2I + I2O:  o_emb = l2norm(lrelu(neigh @ Wo + bo)) ----------------
__global__ __launch_bounds__(256)
void i2i_kernel(const __bf16* __restrict__ itemb, const int* __restrict__ omap,
                const __bf16* __restrict__ W1t, const float* __restrict__ b1,
                const __bf16* __restrict__ W2t, const float* __restrict__ b2,
                const __bf16* __restrict__ Wot, const float* __restrict__ bo,
                float* __restrict__ o_emb) {
    __shared__ __align__(16) char smem[62720];
    __bf16* sP0 = (__bf16*)(smem);                 // 64x136 bf16 (16B-aligned rows)
    __bf16* sP1 = (__bf16*)(smem + 17408);
    __bf16* sNB = (__bf16*)(smem + 34816);         // neigh (bf16)
    __bf16* sB  = (__bf16*)(smem + 52224);         // 128x40 weight K-slice
    float*  sNrm = (float*)(smem + 62464);
    float*  sO   = (float*)(smem);                 // final-phase alias of sP0+sP1 (64x132 f32)

    const int tid = threadIdx.x, lane = tid & 31, wave = tid >> 5;
    const int wr = wave & 3, wc = wave >> 2;       // 16 rows x 64 cols per wave
    const int g = lane >> 4, cidx = lane & 15;
    const int o0 = blockIdx.x * 64;
    const int lrow = tid >> 2, kk = (tid & 3) * 32;
    const int brow = tid >> 1, bhalf = tid & 1;

    {   // async gather of bf16 item features: 4 threads per outfit row, 64B each
        int oc = min(o0 + lrow, NO - 1);
        int ia = omap[2 * oc], ib = omap[2 * oc + 1];
        const __bf16* pa = itemb + (size_t)ia * D + kk;
        const __bf16* pb = itemb + (size_t)ib * D + kk;
        __bf16* d0 = sP0 + lrow * 136 + kk;
        __bf16* d1 = sP1 + lrow * 136 + kk;
#pragma unroll
        for (int q = 0; q < 4; ++q) {
            async_ld_b128(d0 + q * 8, pa + q * 8);
            async_ld_b128(d1 + q * 8, pb + q * 8);
        }
    }
    // -------- neigh = lrelu(P0@W1+b1) + lrelu(P1@W1+b1) + lrelu((P0*P1)@W2+b2) --------
    v8f nacc[4];
#pragma unroll
    for (int j = 0; j < 4; ++j) nacc[j] = (v8f){0,0,0,0,0,0,0,0};

    for (int ps = 0; ps < 3; ++ps) {
        const __bf16* Wt  = (ps < 2) ? W1t : W2t;
        const float*  bia = (ps < 2) ? b1 : b2;
        v8f t[4];
#pragma unroll
        for (int j = 0; j < 4; ++j) t[j] = (v8f){0,0,0,0,0,0,0,0};
        for (int k0 = 0; k0 < D; k0 += 32) {
            {   // B K-slice (128 x 32)
                const __bf16* gsrc = Wt + (size_t)brow * D + k0 + bhalf * 16;
                __bf16* dst = sB + brow * 40 + bhalf * 16;
                async_ld_b128(dst, gsrc);
                async_ld_b128(dst + 8, gsrc + 8);
            }
            wait_async0();
            __syncthreads();
            v16bf a;
            if (ps == 0)      a = load_a_frag(sP0 + wr * 16 * 136 + k0, 136, lane);
            else if (ps == 1) a = load_a_frag(sP1 + wr * 16 * 136 + k0, 136, lane);
            else {
                v16bf a0 = load_a_frag(sP0 + wr * 16 * 136 + k0, 136, lane);
                v16bf a1 = load_a_frag(sP1 + wr * 16 * 136 + k0, 136, lane);
#pragma unroll
                for (int i = 0; i < 16; ++i) a[i] = f2bf(bf2f(a0[i]) * bf2f(a1[i]));
            }
#pragma unroll
            for (int j = 0; j < 4; ++j) {
                v16bf bb = load_b_frag(sB + (wc * 64 + j * 16) * 40, 40, lane);
                t[j] = WMMA_BF16(a, bb, t[j]);
            }
            __syncthreads();
        }
#pragma unroll
        for (int j = 0; j < 4; ++j) {
            float bi = bia[wc * 64 + j * 16 + cidx];
#pragma unroll
            for (int v = 0; v < 8; ++v) nacc[j][v] += lrelu(t[j][v] + bi);
        }
    }
#pragma unroll
    for (int j = 0; j < 4; ++j) {
        int col = wc * 64 + j * 16 + cidx;
#pragma unroll
        for (int v = 0; v < 8; ++v)
            sNB[(wr * 16 + v + 8 * g) * 136 + col] = f2bf(nacc[j][v]);
    }
    __syncthreads();
    // -------- o = lrelu(neigh @ Wo + bo); then l2norm rows --------
    v8f t[4];
#pragma unroll
    for (int j = 0; j < 4; ++j) t[j] = (v8f){0,0,0,0,0,0,0,0};
    for (int k0 = 0; k0 < D; k0 += 32) {
        {
            const __bf16* gsrc = Wot + (size_t)brow * D + k0 + bhalf * 16;
            __bf16* dst = sB + brow * 40 + bhalf * 16;
            async_ld_b128(dst, gsrc);
            async_ld_b128(dst + 8, gsrc + 8);
        }
        wait_async0();
        __syncthreads();
        v16bf a = load_a_frag(sNB + wr * 16 * 136 + k0, 136, lane);
#pragma unroll
        for (int j = 0; j < 4; ++j) {
            v16bf bb = load_b_frag(sB + (wc * 64 + j * 16) * 40, 40, lane);
            t[j] = WMMA_BF16(a, bb, t[j]);
        }
        __syncthreads();
    }
#pragma unroll
    for (int j = 0; j < 4; ++j) {
        int col = wc * 64 + j * 16 + cidx;
        float bias = bo[col];
#pragma unroll
        for (int v = 0; v < 8; ++v)
            sO[(wr * 16 + v + 8 * g) * 132 + col] = lrelu(t[j][v] + bias);
    }
    __syncthreads();
    if (tid < 64) {
        float s = 0.f;
        for (int d = 0; d < D; ++d) { float x = sO[tid * 132 + d]; s += x * x; }
        sNrm[tid] = 1.f / fmaxf(sqrtf(s), 1e-12f);
    }
    __syncthreads();
    if (o0 + lrow < NO) {
        float rn = sNrm[lrow];
#pragma unroll
        for (int e = 0; e < 32; ++e)
            o_emb[(size_t)(o0 + lrow) * D + kk + e] = sO[lrow * 132 + kk + e] * rn;
    }
}

// ---------------- sparse propagation: side[row] += val * ego[col] ----------------
__global__ __launch_bounds__(256)
void spmv_kernel(const int* __restrict__ Lrow, const int* __restrict__ Lcol,
                 const float* __restrict__ Lval, const float* __restrict__ ego,
                 float* __restrict__ side) {
    int gid = blockIdx.x * 256 + threadIdx.x;
    int edge = gid >> 2;
    if (edge >= NNZ_E) return;
    int d0 = (gid & 3) * 32;
    int r = Lrow[edge], c = Lcol[edge];
    float v = Lval[edge];
    const float* es = ego + (size_t)c * D + d0;
    float* dst = side + (size_t)r * D + d0;
#pragma unroll
    for (int e = 0; e < 32; e += 4) {
        float4 x = *(const float4*)(es + e);
        atomicAdd(dst + e + 0, v * x.x);
        atomicAdd(dst + e + 1, v * x.y);
        atomicAdd(dst + e + 2, v * x.z);
        atomicAdd(dst + e + 3, v * x.w);
    }
}

// ------- O2U dense: ego = lrelu(side@W1 + b1 + (ego*side)@W2 + b2); all_emb += l2norm(ego) -------
__global__ __launch_bounds__(256)
void o2u_kernel(const float* __restrict__ side, float* __restrict__ ego,
                const __bf16* __restrict__ W1t, const float* __restrict__ b1,
                const __bf16* __restrict__ W2t, const float* __restrict__ b2,
                float* __restrict__ all_emb) {
    __shared__ __align__(16) char smem[45312];
    __bf16* sA1 = (__bf16*)(smem);                 // side (bf16) 64x136
    __bf16* sA2 = (__bf16*)(smem + 17408);         // ego*side (bf16)
    __bf16* sB  = (__bf16*)(smem + 34816);         // 128x40 weight K-slice
    float*  sNrm = (float*)(smem + 45056);
    float*  sO   = (float*)(smem);                 // alias of sA1+sA2 (64x132 f32)

    const int tid = threadIdx.x, lane = tid & 31, wave = tid >> 5;
    const int wr = wave & 3, wc = wave >> 2;       // 16 rows x 64 cols per wave
    const int g = lane >> 4, cidx = lane & 15;
    const int r0 = blockIdx.x * 64;
    const int lrow = tid >> 2, kk = (tid & 3) * 32;
    const int brow = tid >> 1, bhalf = tid & 1;

    {   // stage A tiles (fp32 -> bf16 + elementwise product)
        size_t srow = (size_t)min(r0 + lrow, NNODES - 1);
#pragma unroll
        for (int e = 0; e < 32; e += 4) {
            float4 sv = *(const float4*)(side + srow * D + kk + e);
            float4 ev = *(const float4*)(ego + srow * D + kk + e);
            __bf16* d1 = sA1 + lrow * 136 + kk + e;
            __bf16* d2 = sA2 + lrow * 136 + kk + e;
            d1[0] = f2bf(sv.x); d1[1] = f2bf(sv.y); d1[2] = f2bf(sv.z); d1[3] = f2bf(sv.w);
            d2[0] = f2bf(ev.x * sv.x); d2[1] = f2bf(ev.y * sv.y);
            d2[2] = f2bf(ev.z * sv.z); d2[3] = f2bf(ev.w * sv.w);
        }
    }
    v8f t[4];
#pragma unroll
    for (int j = 0; j < 4; ++j) t[j] = (v8f){0,0,0,0,0,0,0,0};
    for (int ps = 0; ps < 2; ++ps) {               // accumulate both GEMMs into same acc
        const __bf16* Wt = ps ? W2t : W1t;
        const __bf16* sA = ps ? sA2 : sA1;
        for (int k0 = 0; k0 < D; k0 += 32) {
            {
                const __bf16* gsrc = Wt + (size_t)brow * D + k0 + bhalf * 16;
                __bf16* dst = sB + brow * 40 + bhalf * 16;
                async_ld_b128(dst, gsrc);
                async_ld_b128(dst + 8, gsrc + 8);
            }
            wait_async0();
            __syncthreads();
            v16bf a = load_a_frag(sA + wr * 16 * 136 + k0, 136, lane);
#pragma unroll
            for (int j = 0; j < 4; ++j) {
                v16bf bb = load_b_frag(sB + (wc * 64 + j * 16) * 40, 40, lane);
                t[j] = WMMA_BF16(a, bb, t[j]);
            }
            __syncthreads();
        }
    }
    v8f out[4];
#pragma unroll
    for (int j = 0; j < 4; ++j) {
        int col = wc * 64 + j * 16 + cidx;
        float bi = b1[col] + b2[col];
#pragma unroll
        for (int v = 0; v < 8; ++v) out[j][v] = lrelu(t[j][v] + bi);
    }
    __syncthreads();   // all reads of sA1/sA2 done -> safe to alias sO
#pragma unroll
    for (int j = 0; j < 4; ++j) {
        int col = wc * 64 + j * 16 + cidx;
#pragma unroll
        for (int v = 0; v < 8; ++v)
            sO[(wr * 16 + v + 8 * g) * 132 + col] = out[j][v];
    }
    __syncthreads();
    if (tid < 64) {
        float s = 0.f;
        for (int d = 0; d < D; ++d) { float x = sO[tid * 132 + d]; s += x * x; }
        sNrm[tid] = 1.f / fmaxf(sqrtf(s), 1e-12f);
    }
    __syncthreads();
    if (r0 + lrow < NNODES) {
        float rn = sNrm[lrow];
        size_t rowg = (size_t)(r0 + lrow) * D;
#pragma unroll
        for (int e = 0; e < 32; ++e) {
            float val = sO[lrow * 132 + kk + e];
            ego[rowg + kk + e] = val;
            all_emb[rowg + kk + e] += val * rn;
        }
    }
}

// ---------------- host launcher ----------------
extern "C" void kernel_launch(void* const* d_in, const int* in_sizes, int n_in,
                              void* d_out, int out_size, void* d_ws, size_t ws_size,
                              hipStream_t stream) {
    const float* visual   = (const float*)d_in[0];
    const float* corpus   = (const float*)d_in[1];
    const int*   omap     = (const int*)d_in[2];
    const int*   Lidx     = (const int*)d_in[3];
    const float* Lval     = (const float*)d_in[4];
    const float* mlp_w1   = (const float*)d_in[5];
    const float* mlp_b1   = (const float*)d_in[6];
    const float* mlp_w2   = (const float*)d_in[7];
    const float* mlp_b2   = (const float*)d_in[8];
    const float* i2i_w1   = (const float*)d_in[9];
    const float* i2i_b1   = (const float*)d_in[10];
    const float* i2i_w2   = (const float*)d_in[11];
    const float* i2i_b2   = (const float*)d_in[12];
    const float* i2o_w    = (const float*)d_in[13];
    const float* i2o_b    = (const float*)d_in[14];
    const float* user_emb = (const float*)d_in[15];
    const float* o2u_w1   = (const float*)d_in[16];
    const float* o2u_b1   = (const float*)d_in[17];
    const float* o2u_w2   = (const float*)d_in[18];
    const float* o2u_b2   = (const float*)d_in[19];
    float* out = (float*)d_out;

    char* ws = (char*)d_ws;
    size_t off = 0;
    auto carve = [&](size_t bytes) -> char* {
        char* p = ws + off; off += (bytes + 255) & ~(size_t)255; return p;
    };
    __bf16* wt_m1 = (__bf16*)carve((size_t)DH * DIN * 2);       // W1^T [256][512]
    __bf16* wt_m2 = (__bf16*)carve((size_t)D * DH * 2);         // W2^T [128][256]
    __bf16* wt_i1 = (__bf16*)carve((size_t)D * D * 2);
    __bf16* wt_i2 = (__bf16*)carve((size_t)D * D * 2);
    __bf16* wt_io = (__bf16*)carve((size_t)D * D * 2);
    __bf16* wt_u1 = (__bf16*)carve((size_t)LAYERS * D * D * 2);
    __bf16* wt_u2 = (__bf16*)carve((size_t)LAYERS * D * D * 2);
    __bf16* itemb = (__bf16*)carve((size_t)NI * D * 2);
    float*  ego   = (float*)carve((size_t)NNODES * D * 4);
    float*  side  = (float*)carve((size_t)NNODES * D * 4);
    float*  o_emb = ego + (size_t)NU * D;   // ego = concat(user_emb, o_emb)

    // 1. transpose+convert weights to bf16
    wt_kernel<<<(DIN * DH + 255) / 256, 256, 0, stream>>>(mlp_w1, wt_m1, DIN, DH);
    wt_kernel<<<(DH * D + 255) / 256, 256, 0, stream>>>(mlp_w2, wt_m2, DH, D);
    wt_kernel<<<(D * D + 255) / 256, 256, 0, stream>>>(i2i_w1, wt_i1, D, D);
    wt_kernel<<<(D * D + 255) / 256, 256, 0, stream>>>(i2i_w2, wt_i2, D, D);
    wt_kernel<<<(D * D + 255) / 256, 256, 0, stream>>>(i2o_w, wt_io, D, D);
    for (int l = 0; l < LAYERS; ++l) {
        wt_kernel<<<(D * D + 255) / 256, 256, 0, stream>>>(o2u_w1 + (size_t)l * D * D,
                                                           wt_u1 + (size_t)l * D * D, D, D);
        wt_kernel<<<(D * D + 255) / 256, 256, 0, stream>>>(o2u_w2 + (size_t)l * D * D,
                                                           wt_u2 + (size_t)l * D * D, D, D);
    }
    // 2. MLP embed: item feats (bf16 scratch) + corpus emb (fp32, straight into d_out tail)
    mlp_kernel<<<(NI + 63) / 64, 256, 0, stream>>>(visual, wt_m1, mlp_b1, wt_m2, mlp_b2,
                                                   (void*)itemb, NI, 1);
    mlp_kernel<<<(NCNT + 63) / 64, 256, 0, stream>>>(corpus, wt_m1, mlp_b1, wt_m2, mlp_b2,
                                                     (void*)(out + (size_t)NNODES * D), NCNT, 0);
    // 3. ego = concat(user_emb, o_emb)
    hipMemcpyAsync(ego, user_emb, (size_t)NU * D * 4, hipMemcpyDeviceToDevice, stream);
    i2i_kernel<<<(NO + 63) / 64, 256, 0, stream>>>(itemb, omap, wt_i1, i2i_b1, wt_i2, i2i_b2,
                                                   wt_io, i2o_b, o_emb);
    // 4. all_emb = ego
    hipMemcpyAsync(out, ego, (size_t)NNODES * D * 4, hipMemcpyDeviceToDevice, stream);
    // 5. LightGCN-style layers
    for (int l = 0; l < LAYERS; ++l) {
        zero_kernel<<<(NNODES * D / 4 + 255) / 256, 256, 0, stream>>>(side, NNODES * D / 4);
        spmv_kernel<<<(NNZ_E * 4 + 255) / 256, 256, 0, stream>>>(Lidx, Lidx + NNZ_E, Lval,
                                                                 ego, side);
        o2u_kernel<<<(NNODES + 63) / 64, 256, 0, stream>>>(side, ego,
                                                           wt_u1 + (size_t)l * D * D, o2u_b1 + l * D,
                                                           wt_u2 + (size_t)l * D * D, o2u_b2 + l * D,
                                                           out);
    }
}
